// DLRMResNet_40389872451969
// MI455X (gfx1250) — matrix operand order repaired and checked by reading, not compiled, not run on hardware
//
#include <hip/hip_runtime.h>

typedef unsigned short u16;
typedef __attribute__((ext_vector_type(16))) __bf16 v16bf;
typedef __attribute__((ext_vector_type(8)))  float  v8f;
typedef int v4i __attribute__((__vector_size__(16)));

#define BATCH   16384
#define VOCAB   4194304u
#define CHV     1048576u
#define DEMB    128
#define MTILE   64
#define KC      128
#define LROW    (KC + 8)      // 136 u16 -> 272B rows, breaks bank conflicts
#define AROW    (256 + 8)     // 264
#define AFR     (256 + 4)     // 260

#define GAS __attribute__((address_space(1)))
#define LAS __attribute__((address_space(3)))

__device__ __forceinline__ u16 f2bf(float f) {
  unsigned u = __float_as_uint(f);
  unsigned r = 0x7FFFu + ((u >> 16) & 1u);      // round-to-nearest-even
  return (u16)((u + r) >> 16);
}

__device__ __forceinline__ v8f v8f_zero() {
  v8f z = {0.f, 0.f, 0.f, 0.f, 0.f, 0.f, 0.f, 0.f};
  return z;
}

// 16B global -> LDS copy. Prefer CDNA5 async DMA (ASYNCcnt, bypasses VGPRs);
// fall back to a synchronous register copy if the builtin is unavailable.
__device__ __forceinline__ void async_cp16(u16* lds, const u16* g) {
#if __has_builtin(__builtin_amdgcn_global_load_async_to_lds_b128)
  __builtin_amdgcn_global_load_async_to_lds_b128(
      (GAS v4i*)(g), (LAS v4i*)(lds), 0, 0);
#else
  *(uint4*)lds = *(const uint4*)g;
#endif
}

__device__ __forceinline__ void wait_async0() {
#if __has_builtin(__builtin_amdgcn_global_load_async_to_lds_b128)
#if __has_builtin(__builtin_amdgcn_s_wait_asynccnt)
  __builtin_amdgcn_s_wait_asynccnt(0);
#else
  asm volatile("s_wait_asynccnt 0x0" ::: "memory");
#endif
#endif
}

// A fragment: 16x32 bf16, ISA layout: lanes0-15 M=lane, VGPR i holds K-pair
// k = (i>>2)*16 + (lane>>4)*8 + (i&3)*2
__device__ __forceinline__ v16bf load_afrag(const u16* A, int SA, int m0, int kbase, int lane) {
  union { unsigned u[8]; v16bf v; } f;
  const u16* row = A + (m0 + (lane & 15)) * SA + kbase;
  const int half8 = (lane >> 4) << 3;
  #pragma unroll
  for (int i = 0; i < 8; ++i) {
    int k = ((i >> 2) << 4) + half8 + ((i & 3) << 1);
    f.u[i] = *(const unsigned*)(row + k);
  }
  return f.v;
}

// B fragment: 32x16 bf16 from transposed ([N][K]) weights in LDS:
// VGPR i packs K-pair k = (i>>2)*16 + (lane>>4)*8 + (i&3)*2 at N=lane&15
__device__ __forceinline__ v16bf load_bfrag(const u16* Bt, int SB, int n0, int kbase, int lane) {
  union { unsigned u[8]; v16bf v; } f;
  const u16* row = Bt + (n0 + (lane & 15)) * SB + kbase + ((lane >> 4) << 3);
  #pragma unroll
  for (int i = 0; i < 8; ++i) f.u[i] = *(const unsigned*)(row + ((i >> 2) << 4) + ((i & 3) << 1));
  return f.v;
}

__device__ __forceinline__ v8f wmma_bf16(v16bf a, v16bf b, v8f c) {
  return __builtin_amdgcn_wmma_f32_16x16x32_bf16(false, a, false, b, (short)0, c, false, false);
}

// Residual layer: act(64x256) = relu(act @ W(256x256) + b) + act, all in LDS.
// Whole W^T async-staged into both halves of sw, then 8 uninterrupted K-steps.
__device__ __forceinline__ void res_layer256(
    const u16* __restrict__ wt, const float* __restrict__ bias,
    u16 (*sw)[256][LROW], u16 (*sabf)[AROW], float (*saf)[AFR], int tid)
{
  const int lane = tid & 31, wid = tid >> 5;
  const int mi = wid & 3, nh = wid >> 2;
  #pragma unroll
  for (int c = 0; c < 2; ++c) {
    const u16* src = wt + tid * 256 + c * KC;
    #pragma unroll
    for (int i = 0; i < 16; ++i) async_cp16(&sw[c][tid][i * 8], src + i * 8);
  }
  wait_async0();
  __syncthreads();

  v8f acc[8];
  #pragma unroll
  for (int t = 0; t < 8; ++t) acc[t] = v8f_zero();
  #pragma unroll
  for (int c = 0; c < 2; ++c) {
    #pragma unroll
    for (int ks = 0; ks < 4; ++ks) {
      v16bf a = load_afrag(&sabf[0][0], AROW, mi * 16, c * KC + ks * 32, lane);
      #pragma unroll
      for (int t = 0; t < 8; ++t) {
        v16bf b = load_bfrag(&sw[c][0][0], LROW, nh * 128 + t * 16, ks * 32, lane);
        acc[t] = wmma_bf16(a, b, acc[t]);
      }
    }
  }
  __syncthreads();
  const int n_ = lane & 15, mb = (lane >> 4) << 3;
  #pragma unroll
  for (int t = 0; t < 8; ++t) {
    int col = nh * 128 + t * 16 + n_;
    float bv = bias[col];
    #pragma unroll
    for (int v = 0; v < 8; ++v) {
      int m = mi * 16 + mb + v;
      float val = fmaxf(acc[t][v] + bv, 0.f) + saf[m][col];
      saf[m][col] = val;
      sabf[m][col] = f2bf(val);
    }
  }
  __syncthreads();
}

// ---------------- bottom MLP: 13 -> 256 -> 256 -> 256 ----------------
__global__ __launch_bounds__(256, 1) void k_bottom(
    const float* __restrict__ x, const float* __restrict__ bw0, const float* __restrict__ bb0,
    const u16* __restrict__ bw1t, const float* __restrict__ bb1,
    const u16* __restrict__ bw2t, const float* __restrict__ bb2,
    u16* __restrict__ hbuf)
{
  __shared__ u16   sw[2][256][LROW];
  __shared__ u16   sabf[MTILE][AROW];
  __shared__ float saf[MTILE][AFR];
  const int tid = threadIdx.x;
  const int row0 = blockIdx.x * MTILE;

  {   // layer 0, K=13, plain f32 VALU
    int r = tid & 63, cg = tid >> 6;
    float d[13];
    #pragma unroll
    for (int k = 0; k < 13; ++k) d[k] = x[(row0 + r) * 39 + k];
    for (int j = 0; j < 64; ++j) {
      int n = cg * 64 + j;
      float s = bb0[n];
      #pragma unroll
      for (int k = 0; k < 13; ++k) s = fmaf(d[k], bw0[k * 256 + n], s);
      s = fmaxf(s, 0.f);
      saf[r][n] = s;
      sabf[r][n] = f2bf(s);
    }
  }
  __syncthreads();
  res_layer256(bw1t, bb1, sw, sabf, saf, tid);
  res_layer256(bw2t, bb2, sw, sabf, saf, tid);
  {   // coalesced bf16 writeback of h
    int r = tid >> 2, q = tid & 3;
    const uint4* src = (const uint4*)&sabf[r][q * 64];
    uint4* dst = (uint4*)(hbuf + (size_t)(row0 + r) * 256 + q * 64);
    #pragma unroll
    for (int i = 0; i < 8; ++i) dst[i] = src[i];
  }
}

// ------------- fused top MLP: gather + 3584->256 ->256->256->256 ->1 -------------
__global__ __launch_bounds__(256, 1) void k_top(
    const float* __restrict__ x,
    const float* __restrict__ e0, const float* __restrict__ e1,
    const float* __restrict__ e2, const float* __restrict__ e3,
    const u16* __restrict__ hbuf,
    const u16* __restrict__ tw0t, const float* __restrict__ tb0,
    const u16* __restrict__ tw1t, const float* __restrict__ tb1,
    const u16* __restrict__ tw2t, const float* __restrict__ tb2,
    const u16* __restrict__ tw3t, const float* __restrict__ tb3,
    const float* __restrict__ tw4, const float* __restrict__ tb4,
    float* __restrict__ out)
{
  __shared__ u16   sw[2][256][LROW];   // double-buffered weight chunk (2 x 64KB)
  __shared__ u16   sz[2][MTILE][LROW]; // double-buffered z chunk
  __shared__ u16   sabf[MTILE][AROW];
  __shared__ float saf[MTILE][AFR];
  const int tid = threadIdx.x;
  const int lane = tid & 31, wid = tid >> 5;
  const int mi = wid & 3, nh = wid >> 2;
  const int row0 = blockIdx.x * MTILE;
  const float* etab[4] = {e0, e1, e2, e3};

  v8f acc[8];
  #pragma unroll
  for (int t = 0; t < 8; ++t) acc[t] = v8f_zero();

  const int r = tid >> 2, q = tid & 3;   // staging: row, 32-col quarter
  float4 pf[8];                          // embedding prefetch registers

  // issue chunk c's staging into buffer b (async DMA for bf16 bytes,
  // register prefetch for the f32 embedding rows that need conversion)
  auto issue_stage = [&](int c, int b) {
    const u16* wsrc = tw0t + tid * 3584 + c * KC;
    #pragma unroll
    for (int i = 0; i < 16; ++i) async_cp16(&sw[b][tid][i * 8], wsrc + i * 8);
    if (c < 2) {                         // h columns: pure byte copy, async
      const u16* hsrc = hbuf + (size_t)(row0 + r) * 256 + c * KC + q * 32;
      #pragma unroll
      for (int i = 0; i < 4; ++i) async_cp16(&sz[b][r][q * 32 + i * 8], hsrc + i * 8);
    } else {                             // embedding slot: prefetch f32 row piece
      int s = c - 2;
      float fi = x[(row0 + r) * 39 + 13 + s];
      unsigned idx = ((unsigned)(int)fi) & (VOCAB - 1u);
      const float* trow = etab[idx >> 20] + (size_t)(idx & (CHV - 1u)) * DEMB + q * 32;
      #pragma unroll
      for (int i = 0; i < 8; ++i) pf[i] = ((const float4*)trow)[i];
    }
  };
  // complete chunk c's staging (convert + LDS store, drain async DMA)
  auto finish_stage = [&](int c, int b) {
    if (c >= 2) {
      u16* dst = &sz[b][r][q * 32];
      #pragma unroll
      for (int i = 0; i < 8; ++i) {
        dst[i * 4 + 0] = f2bf(pf[i].x); dst[i * 4 + 1] = f2bf(pf[i].y);
        dst[i * 4 + 2] = f2bf(pf[i].z); dst[i * 4 + 3] = f2bf(pf[i].w);
      }
    }
    wait_async0();
  };

  // ---- layer 0 over z = [h | emb], K = 3584 = 28 chunks, double buffered ----
  issue_stage(0, 0);
  finish_stage(0, 0);
  __syncthreads();
  for (int c = 0; c < 28; ++c) {
    const int cur = c & 1, nxt = cur ^ 1;
    if (c < 27) issue_stage(c + 1, nxt);     // overlap staging with WMMA
    #pragma unroll
    for (int ks = 0; ks < 4; ++ks) {
      v16bf a = load_afrag(&sz[cur][0][0], LROW, mi * 16, ks * 32, lane);
      #pragma unroll
      for (int t = 0; t < 8; ++t) {
        v16bf b = load_bfrag(&sw[cur][0][0], LROW, nh * 128 + t * 16, ks * 32, lane);
        acc[t] = wmma_bf16(a, b, acc[t]);
      }
    }
    if (c < 27) finish_stage(c + 1, nxt);
    __syncthreads();
  }
  {   // layer-0 epilogue: bias + relu, no residual
    const int n_ = lane & 15, mb = (lane >> 4) << 3;
    #pragma unroll
    for (int t = 0; t < 8; ++t) {
      int col = nh * 128 + t * 16 + n_;
      float bv = tb0[col];
      #pragma unroll
      for (int v = 0; v < 8; ++v) {
        int m = mi * 16 + mb + v;
        float val = fmaxf(acc[t][v] + bv, 0.f);
        saf[m][col] = val;
        sabf[m][col] = f2bf(val);
      }
    }
  }
  __syncthreads();
  res_layer256(tw1t, tb1, sw, sabf, saf, tid);
  res_layer256(tw2t, tb2, sw, sabf, saf, tid);
  res_layer256(tw3t, tb3, sw, sabf, saf, tid);
  {   // final 256 -> 1: 4 threads per row + wave32 shuffle reduction
    int rr = tid >> 2, qq = tid & 3;
    float sum = 0.f;
    #pragma unroll
    for (int k = 0; k < 64; ++k) sum = fmaf(saf[rr][qq * 64 + k], tw4[qq * 64 + k], sum);
    sum += __shfl_xor(sum, 1, 32);
    sum += __shfl_xor(sum, 2, 32);
    if (qq == 0) out[row0 + rr] = sum + tb4[0];
  }
}

// transpose-convert f32 [K][N] -> bf16 [N][K]
__global__ void k_convert_t(const float* __restrict__ src, u16* __restrict__ dst,
                            int K, int N, int total) {
  int i = blockIdx.x * blockDim.x + threadIdx.x;
  if (i >= total) return;
  int n = i / K, k = i - n * K;
  dst[i] = f2bf(src[(size_t)k * N + n]);
}

extern "C" void kernel_launch(void* const* d_in, const int* in_sizes, int n_in,
                              void* d_out, int out_size, void* d_ws, size_t ws_size,
                              hipStream_t stream) {
  (void)in_sizes; (void)n_in; (void)out_size; (void)ws_size;
  const float* x  = (const float*)d_in[0];
  const float* e0 = (const float*)d_in[1];
  const float* e1 = (const float*)d_in[2];
  const float* e2 = (const float*)d_in[3];
  const float* e3 = (const float*)d_in[4];

  char* ws = (char*)d_ws;
  u16* hbuf = (u16*)ws; ws += (size_t)BATCH * 256 * 2;
  u16* bw1t = (u16*)ws; ws += 256 * 256 * 2;
  u16* bw2t = (u16*)ws; ws += 256 * 256 * 2;
  u16* tw0t = (u16*)ws; ws += (size_t)256 * 3584 * 2;
  u16* tw1t = (u16*)ws; ws += 256 * 256 * 2;
  u16* tw2t = (u16*)ws; ws += 256 * 256 * 2;
  u16* tw3t = (u16*)ws; ws += 256 * 256 * 2;

  auto conv = [&](const void* s, u16* d, int K, int N) {
    int total = K * N;
    k_convert_t<<<(total + 255) / 256, 256, 0, stream>>>((const float*)s, d, K, N, total);
  };
  conv(d_in[7],  bw1t, 256, 256);
  conv(d_in[9],  bw2t, 256, 256);
  conv(d_in[11], tw0t, 3584, 256);
  conv(d_in[13], tw1t, 256, 256);
  conv(d_in[15], tw2t, 256, 256);
  conv(d_in[17], tw3t, 256, 256);

  k_bottom<<<BATCH / MTILE, 256, 0, stream>>>(
      x, (const float*)d_in[5], (const float*)d_in[6],
      bw1t, (const float*)d_in[8], bw2t, (const float*)d_in[10], hbuf);

  k_top<<<BATCH / MTILE, 256, 0, stream>>>(
      x, e0, e1, e2, e3, hbuf,
      tw0t, (const float*)d_in[12],
      tw1t, (const float*)d_in[14],
      tw2t, (const float*)d_in[16],
      tw3t, (const float*)d_in[18],
      (const float*)d_in[19], (const float*)d_in[20],
      (float*)d_out);
}
